// SelfMultiheadAttention_21715354649691
// MI455X (gfx1250) — matrix operand order repaired and checked by reading, not compiled, not run on hardware
//
#include <hip/hip_runtime.h>
#include <hip/hip_bf16.h>

// ---------------------------------------------------------------------------
// SelfMultiheadAttention for MI455X (gfx1250), wave32, BF16 WMMA (f32 accum).
//   B=4, T=1024, E=1024, H=16, D=64
// Roofline: 52 GFLOP total; mandatory HBM traffic dominated by attn_bias
// (256 MB -> ~11us at 23.3 TB/s). Weights+activations fit in 192MB L2, so
// GEMM fragments stream from global (L2) directly. The per-head V slab is
// DMA'd into LDS by the Tensor Data Mover (overlapped with QK^T + softmax)
// so the transpose-side reads of P@V hit LDS, not global.
// ---------------------------------------------------------------------------

typedef __bf16 bf16;
typedef __attribute__((ext_vector_type(16))) __bf16 v16bf;
typedef __attribute__((ext_vector_type(8)))  float  v8f;
typedef unsigned int v4u __attribute__((ext_vector_type(4)));
typedef int          v8i __attribute__((ext_vector_type(8)));
typedef int          v4i __attribute__((ext_vector_type(4)));

struct __align__(16) U128 { unsigned int w[4]; };

#define Bc 4
#define Tc 1024
#define Ec 1024
#define Hc 16
#define Dc 64
#define E3 (3 * Ec)

__device__ __forceinline__ v8f wmma_bf16(v16bf a, v16bf b, v8f c) {
    // v_wmma_f32_16x16x32_bf16  D = A(16x32) * B(32x16) + C(16x16 f32)
    return __builtin_amdgcn_wmma_f32_16x16x32_bf16(
        /*neg_a=*/false, a, /*neg_b=*/false, b,
        /*c_mod=*/(short)0, c, /*reuse_a=*/false, /*reuse_b=*/false);
}

// A-matrix 16x32 bf16 fragment (ISA 7.12.2 layout):
// lane<16 : row M=lane,    K = {0..7, 16..23} ; lane>=16: K = {8..15, 24..31}
__device__ __forceinline__ v16bf load_a_frag(const bf16* base, int ld) {
    int lane = threadIdx.x & 31;
    int m    = lane & 15;
    int k0   = (lane >> 4) ? 8 : 0;
    const bf16* p = base + (size_t)m * ld + k0;
    v16bf r;
    ((U128*)&r)[0] = *(const U128*)(p);
    ((U128*)&r)[1] = *(const U128*)(p + 16);
    return r;
}

// B-matrix 32x16 bf16 fragment where column n of B == contiguous row n of the
// source (covers W[N,K] row-major and K-matrix [keys, D] row-major):
// lane<16 : col N=lane, K = {0..7, 16..23} ; lane>=16: col N=lane-16, K = {8..15, 24..31}
__device__ __forceinline__ v16bf load_b_frag_rows(const bf16* base, int ld) {
    int lane = threadIdx.x & 31;
    int n    = lane & 15;
    int k0   = (lane >> 4) ? 8 : 0;
    const bf16* p = base + (size_t)n * ld + k0;
    v16bf r;
    ((U128*)&r)[0] = *(const U128*)(p);
    ((U128*)&r)[1] = *(const U128*)(p + 16);
    return r;
}

// B-matrix 32x16 fragment with element B[k][n] = base[k*ld + n] (strided; used
// for V column reads out of LDS).
__device__ __forceinline__ v16bf load_b_frag_strided(const bf16* base, int ld) {
    int lane = threadIdx.x & 31;
    int n    = lane & 15;
    int k0   = (lane >> 4) ? 8 : 0;
    v16bf r;
#pragma unroll
    for (int i = 0; i < 16; ++i) r[i] = base[(size_t)(k0 + i + (i >= 8 ? 8 : 0)) * ld + n];
    return r;
}

// ---------------------------------------------------------------------------
// elementwise f32 -> bf16 cast
// ---------------------------------------------------------------------------
__global__ void cast_f32_bf16_kernel(const float* __restrict__ x,
                                     bf16* __restrict__ y, int n) {
    int i = blockIdx.x * blockDim.x + threadIdx.x;
    if (i < n) y[i] = (bf16)x[i];
}

// ---------------------------------------------------------------------------
// GEMM: C[M,N] = A[M,K](bf16) * W[N,K]^T(bf16) + bias[N](f32)
// 256 threads = 8 waves arranged 4(M) x 2(N); block tile 128x64;
// wave tile 32x32 = 2x2 WMMA tiles; K-step 32.  All operands are L2-resident
// on MI455X (192MB L2), so fragments stream straight from global.
// ---------------------------------------------------------------------------
template <bool OUT_F32>
__global__ void gemm_bf16_kernel(const bf16* __restrict__ A,
                                 const bf16* __restrict__ W,
                                 const float* __restrict__ bias,
                                 void* __restrict__ Cout,
                                 int M, int N, int K) {
    const int lane = threadIdx.x & 31;
    const int wave = threadIdx.x >> 5;
    const int wm   = wave >> 1;   // 0..3
    const int wn   = wave & 1;    // 0..1
    const int m0   = blockIdx.y * 128 + wm * 32;
    const int n0   = blockIdx.x * 64  + wn * 32;

    v8f acc00 = {0.f,0.f,0.f,0.f,0.f,0.f,0.f,0.f};
    v8f acc01 = acc00, acc10 = acc00, acc11 = acc00;

    for (int k = 0; k < K; k += 32) {
        v16bf a0 = load_a_frag(A + (size_t)m0 * K + k, K);
        v16bf a1 = load_a_frag(A + (size_t)(m0 + 16) * K + k, K);
        v16bf b0 = load_b_frag_rows(W + (size_t)n0 * K + k, K);
        v16bf b1 = load_b_frag_rows(W + (size_t)(n0 + 16) * K + k, K);
        acc00 = wmma_bf16(a0, b0, acc00);
        acc01 = wmma_bf16(a0, b1, acc01);
        acc10 = wmma_bf16(a1, b0, acc10);
        acc11 = wmma_bf16(a1, b1, acc11);
    }

    // C layout: vgpr r, lane l -> M = r + 8*(l>>4), N = l&15
    const int n  = lane & 15;
    const int hi = lane >> 4;
#pragma unroll
    for (int ti = 0; ti < 2; ++ti) {
#pragma unroll
        for (int tj = 0; tj < 2; ++tj) {
            const v8f& c = ti ? (tj ? acc11 : acc10) : (tj ? acc01 : acc00);
            int col = n0 + tj * 16 + n;
            float bv = bias[col];
#pragma unroll
            for (int r = 0; r < 8; ++r) {
                int row = m0 + ti * 16 + r + 8 * hi;
                float v = c[r] + bv;
                if (OUT_F32) ((float*)Cout)[(size_t)row * N + col] = v;
                else         ((bf16*)Cout)[(size_t)row * N + col] = (bf16)v;
            }
        }
    }
}

// ---------------------------------------------------------------------------
// Attention: one block per (b, h, 16-query-row tile). 256 threads = 8 waves.
// Entry  : wave 0 kicks a TDM DMA of the whole per-head V slab [1024 x 64]
//          (global row stride 3E) into LDS with a padded pitch of 72 bf16 --
//          this overlaps stages 1-2 entirely (TENSORcnt tracked).
// Stage 1: scores = scale*Q@K^T (+mask,+bias) -> LDS f32 [16 x 1024]
// Stage 2: row softmax (shfl_xor width-16 reductions) -> LDS bf16 P
// Stage 3: O = P@V via WMMA; A-frags from LDS P, B-frags strided from LDS V;
//          waves split 4 d-groups x 2 K-halves, LDS-reduce, bf16 O out.
// ---------------------------------------------------------------------------
#define V_PITCH 72     // bf16 pitch of V in LDS (TDM pads 4 dwords / 32 dwords)
#define S_PITCH 1032   // f32 pitch (padded)
#define P_PITCH 1032   // bf16 pitch (padded, 16B-aligned rows)
#define SMEM_V_BYTES (Tc * V_PITCH * 2)          // 147456
#define SMEM_S_BYTES (16 * S_PITCH * 4)          //  66048
#define SMEM_P_BYTES (16 * P_PITCH * 2)          //  33024
#define SMEM_V_OFF   0
#define SMEM_S_OFF   (SMEM_V_OFF + SMEM_V_BYTES)
#define SMEM_P_OFF   (SMEM_S_OFF + SMEM_S_BYTES)
#define SMEM_BYTES   (SMEM_V_BYTES + SMEM_S_BYTES + SMEM_P_BYTES)

__global__ void attention_kernel(const bf16* __restrict__ qkv,   // [B*T, 3E]
                                 const float* __restrict__ bias, // [B*H, T, T]
                                 const unsigned char* __restrict__ mask, // [B,T]
                                 bf16* __restrict__ obf) {       // [B*T, E]
    extern __shared__ __align__(16) char smem[];
    bf16*  sV = (bf16*)(smem + SMEM_V_OFF);         // [1024][V_PITCH]
    float* sS = (float*)(smem + SMEM_S_OFF);        // [16][S_PITCH]
    bf16*  sP = (bf16*)(smem + SMEM_P_OFF);         // [16][P_PITCH]

    const int q0   = blockIdx.x * 16;
    const int bh   = blockIdx.y;           // b*H + h
    const int b    = bh / Hc;
    const int h    = bh % Hc;
    const int lane = threadIdx.x & 31;
    const int wave = threadIdx.x >> 5;
    const float SCALE = 0.125f;            // 1/sqrt(D), D = 64

    // ---- TDM: async DMA V[1024 x 64] (stride 3E) -> LDS, padded pitch 72 ----
    if (wave == 0) {
        const bf16* Vglob = qkv + (size_t)(b * Tc) * E3 + 2 * Ec + h * Dc;
        unsigned long long ga = (unsigned long long)(size_t)Vglob;
        v4u g0;
        g0.x = 1u;                                   // count=1, user mode, no gather
        g0.y = (unsigned int)SMEM_V_OFF;             // lds_addr (bytes)
        g0.z = (unsigned int)ga;                     // global_addr[31:0]
        g0.w = (unsigned int)(ga >> 32) | (2u << 30);// global_addr[56:32] | type=2
        v8i g1;
        g1[0] = (1 << 16)        // data_size = 2 bytes
              | (1 << 20)        // pad_enable
              | (4 << 22)        // pad_interval = 32 dwords (one 64-elem row)
              | (3 << 25);       // pad_amount  = 4 dwords (8 bf16) -> pitch 72
        g1[1] = (64 << 16);      // tensor_dim0 = 64  (bits 79:48, low half here)
        g1[2] = (1024 << 16);    // tensor_dim1 = 1024 (bits 111:80)
        g1[3] = (64 << 16);      // tile_dim0 = 64 (bits 127:112)
        g1[4] = 1024;            // tile_dim1 = 1024 (bits 143:128), tile_dim2=0
        g1[5] = E3;              // tensor_dim0_stride = 3072 (bits 207:160)
        g1[6] = 0;
        g1[7] = 0;
        v4i gz = {0, 0, 0, 0};
#if defined(__clang_major__) && (__clang_major__ >= 23)
        v8i gz8 = {0, 0, 0, 0, 0, 0, 0, 0};
        __builtin_amdgcn_tensor_load_to_lds(g0, g1, gz, gz, gz8, 0);
#else
        __builtin_amdgcn_tensor_load_to_lds(g0, g1, gz, gz, 0);
#endif
    }

    // ---- Stage 1: scores (overlaps the TDM DMA) ----
    const bf16* Qbase = qkv + (size_t)(b * Tc + q0) * E3 + h * Dc;
    const bf16* Kbase = qkv + (size_t)(b * Tc) * E3 + Ec + h * Dc;
    v16bf aq0 = load_a_frag(Qbase, E3);        // D 0..31
    v16bf aq1 = load_a_frag(Qbase + 32, E3);   // D 32..63

    const int n  = lane & 15;
    const int hi = lane >> 4;
    for (int nt = 0; nt < 8; ++nt) {
        const int key0 = wave * 128 + nt * 16;
        v8f acc = {0.f,0.f,0.f,0.f,0.f,0.f,0.f,0.f};
        v16bf bk0 = load_b_frag_rows(Kbase + (size_t)key0 * E3, E3);
        v16bf bk1 = load_b_frag_rows(Kbase + (size_t)key0 * E3 + 32, E3);
        acc = wmma_bf16(aq0, bk0, acc);
        acc = wmma_bf16(aq1, bk1, acc);

        const int key = key0 + n;
        const bool km = mask[b * Tc + key] != 0;
        const float* bp = bias + ((size_t)bh * Tc + q0 + hi * 8) * Tc + key;
        // prefetch next tile's bias column (global_prefetch_b8)
        __builtin_prefetch(bp + 16, 0, 0);
#pragma unroll
        for (int r = 0; r < 8; ++r) {
            float s = acc[r] * SCALE;
            if (km) s = -3.0e38f;
            s += bp[(size_t)r * Tc];
            sS[(r + hi * 8) * S_PITCH + key] = s;
        }
    }
    __syncthreads();

    // ---- Stage 2: softmax (16 threads per row) ----
    {
        const int r  = threadIdx.x >> 4;
        const int c0 = threadIdx.x & 15;
        float mx = -3.4e38f;
        for (int j = c0; j < Tc; j += 16) mx = fmaxf(mx, sS[r * S_PITCH + j]);
#pragma unroll
        for (int off = 8; off >= 1; off >>= 1) mx = fmaxf(mx, __shfl_xor(mx, off, 16));
        float sum = 0.f;
        for (int j = c0; j < Tc; j += 16) {
            float e = __expf(sS[r * S_PITCH + j] - mx);
            sS[r * S_PITCH + j] = e;
            sum += e;
        }
#pragma unroll
        for (int off = 8; off >= 1; off >>= 1) sum += __shfl_xor(sum, off, 16);
        const float inv = 1.0f / sum;
        for (int j = c0; j < Tc; j += 16)
            sP[r * P_PITCH + j] = (bf16)(sS[r * S_PITCH + j] * inv);
    }

    // V DMA must be complete before stage 3 consumes LDS V.
    if (wave == 0) __builtin_amdgcn_s_wait_tensorcnt(0);
    __syncthreads();

    // ---- Stage 3: O = P @ V (all operands in LDS) ----
    const int cg = wave & 3;   // d column group (16 cols each)
    const int kh = wave >> 2;  // key half (512 keys each)

    v8f acc = {0.f,0.f,0.f,0.f,0.f,0.f,0.f,0.f};
    for (int step = 0; step < 16; ++step) {
        const int k0 = kh * 512 + step * 32;
        // A fragment from LDS P (contiguous per the A layout)
        const int m  = lane & 15;
        const bf16* pa = sP + m * P_PITCH + k0 + hi * 8;
        v16bf a;
        ((U128*)&a)[0] = *(const U128*)(pa);
        ((U128*)&a)[1] = *(const U128*)(pa + 16);
        // B fragment: strided column reads from padded LDS V
        v16bf bv = load_b_frag_strided(sV + (size_t)k0 * V_PITCH + cg * 16, V_PITCH);
        acc = wmma_bf16(a, bv, acc);
    }

    float* red = sS;  // reuse score LDS as reduction area: [cg][m][n]
    if (kh == 1) {
#pragma unroll
        for (int r = 0; r < 8; ++r)
            red[(cg * 16 + r + 8 * hi) * 16 + n] = acc[r];
    }
    __syncthreads();
    if (kh == 0) {
#pragma unroll
        for (int r = 0; r < 8; ++r) {
            const int m = r + 8 * hi;
            float v = acc[r] + red[(cg * 16 + m) * 16 + n];
            obf[(size_t)(b * Tc + q0 + m) * Ec + h * Dc + cg * 16 + n] = (bf16)v;
        }
    }
}

// ---------------------------------------------------------------------------
// launch
// ---------------------------------------------------------------------------
extern "C" void kernel_launch(void* const* d_in, const int* in_sizes, int n_in,
                              void* d_out, int out_size, void* d_ws, size_t ws_size,
                              hipStream_t stream) {
    const float*         query  = (const float*)d_in[0];
    const float*         abias  = (const float*)d_in[1];
    const unsigned char* mask   = (const unsigned char*)d_in[2]; // jnp.bool_ (1 byte)
    const float*         in_w   = (const float*)d_in[3];
    const float*         in_b   = (const float*)d_in[4];
    const float*         out_w  = (const float*)d_in[5];
    const float*         out_b  = (const float*)d_in[6];
    float*               out    = (float*)d_out;

    // workspace layout (bytes)
    char* ws = (char*)d_ws;
    bf16* qbf    = (bf16*)(ws + 0);          // B*T*E     bf16 =  8,388,608 B
    bf16* winbf  = (bf16*)(ws + 8388608);    // 3E*E      bf16 =  6,291,456 B
    bf16* woutbf = (bf16*)(ws + 14680064);   // E*E       bf16 =  2,097,152 B
    bf16* qkvbf  = (bf16*)(ws + 16777216);   // B*T*3E    bf16 = 25,165,824 B
    bf16* obf    = (bf16*)(ws + 41943040);   // B*T*E     bf16 =  8,388,608 B

    // 1) casts to bf16
    {
        int nq = Bc * Tc * Ec;       // 4,194,304
        int nw = 3 * Ec * Ec;        // 3,145,728
        int no = Ec * Ec;            // 1,048,576
        cast_f32_bf16_kernel<<<nq / 256, 256, 0, stream>>>(query, qbf, nq);
        cast_f32_bf16_kernel<<<nw / 256, 256, 0, stream>>>(in_w, winbf, nw);
        cast_f32_bf16_kernel<<<no / 256, 256, 0, stream>>>(out_w, woutbf, no);
    }

    // 2) QKV projection: [4096,1024] x [3072,1024]^T -> bf16 [4096,3072]
    gemm_bf16_kernel<false><<<dim3(E3 / 64, (Bc * Tc) / 128), 256, 0, stream>>>(
        qbf, winbf, in_b, qkvbf, Bc * Tc, E3, Ec);

    // 3) attention: grid (T/16 q-tiles, B*H heads)
    attention_kernel<<<dim3(Tc / 16, Bc * Hc), 256, SMEM_BYTES, stream>>>(
        qkvbf, abias, mask, obf);

    // 4) output projection: [4096,1024] x [1024,1024]^T + out_b -> f32 d_out
    gemm_bf16_kernel<true><<<dim3(Ec / 64, (Bc * Tc) / 128), 256, 0, stream>>>(
        obf, woutbf, out_b, out, Bc * Tc, Ec, Ec);
}
